// SBMAttention_87247965651106
// MI455X (gfx1250) — matrix operand order repaired and checked by reading, not compiled, not run on hardware
//
#include <hip/hip_runtime.h>
#include <hip/hip_bf16.h>
#include <stdint.h>

// ---------------------------------------------------------------------------
// Problem constants (reference shapes)
// ---------------------------------------------------------------------------
#define Bc 2
#define Hc 8
#define Nc 2048
#define Mc 2048
#define Dc 64
#define BHc (Bc*Hc)
#define SC_STRIDE 2052   // 2052 % 64 == 4 -> spreads LDS banks across rows

static constexpr long long XSZ = (long long)BHc * Nc * Dc;            // 2,097,152
static constexpr long long GSZ = (long long)BHc * Nc * (long long)Mc; // 67,108,864

// ---------------------------------------------------------------------------
// Types / helpers
// ---------------------------------------------------------------------------
typedef __attribute__((ext_vector_type(16))) __bf16       v16bf;
typedef __attribute__((ext_vector_type(8)))  float        v8f;
typedef __attribute__((ext_vector_type(4)))  float        v4f;
typedef __attribute__((ext_vector_type(4)))  unsigned int v4u;

union ABPack { v4u q[2]; v16bf v; };

__device__ __forceinline__ unsigned short f2bf(float f){
  unsigned int u = __float_as_uint(f);
  u += 0x7FFFu + ((u >> 16) & 1u);          // round-to-nearest-even
  return (unsigned short)(u >> 16);
}
__device__ __forceinline__ v4u pack8(v4f a, v4f b){
  v4u r;
  r.x = (unsigned)f2bf(a.x) | ((unsigned)f2bf(a.y) << 16);
  r.y = (unsigned)f2bf(a.z) | ((unsigned)f2bf(a.w) << 16);
  r.z = (unsigned)f2bf(b.x) | ((unsigned)f2bf(b.y) << 16);
  r.w = (unsigned)f2bf(b.z) | ((unsigned)f2bf(b.w) << 16);
  return r;
}
// A-operand (16-bit A 16x32): lane holds row L%16; elements K = off + {0..7, 16..23}
__device__ __forceinline__ v16bf load_bf_A(const unsigned short* row, int off){
  ABPack p;
  p.q[0] = *(const v4u*)(row + off);
  p.q[1] = *(const v4u*)(row + off + 16);
  return p.v;
}
// B-operand (16-bit B 32x16): lane holds col L%16; elements K = off + {0..7, 16..23}
__device__ __forceinline__ v16bf load_bf_B(const unsigned short* row, int off){
  ABPack p;
  p.q[0] = *(const v4u*)(row + off);
  p.q[1] = *(const v4u*)(row + off + 16);
  return p.v;
}

// ---------------------------------------------------------------------------
// Kernel: bf16 conversion prep (Q,K row-major; V transposed [bh,d,M]) + acc=0
// ---------------------------------------------------------------------------
__global__ void k_prep(const float* __restrict__ Q, const float* __restrict__ K,
                       const float* __restrict__ V,
                       unsigned short* __restrict__ Qbf, unsigned short* __restrict__ Kbf,
                       unsigned short* __restrict__ Vt, float* __restrict__ acc){
  long long idx = (long long)blockIdx.x * blockDim.x + threadIdx.x;
  if (idx < 8) acc[idx] = 0.f;
  if (idx >= XSZ) return;
  Qbf[idx] = f2bf(Q[idx]);
  Kbf[idx] = f2bf(K[idx]);
  int bh  = (int)(idx / (Dc * Mc));
  int rem = (int)(idx % (Dc * Mc));
  int d = rem / Mc, m = rem % Mc;
  Vt[idx] = f2bf(V[((long long)bh * Mc + m) * Dc + d]);
}

// ---------------------------------------------------------------------------
// Kernel: S = softmax(clusters @ clusters^T) per head (one wave per head)
// ---------------------------------------------------------------------------
__global__ void k_clusterS(const float* __restrict__ emb, float* __restrict__ S){
  int h = threadIdx.x >> 5;
  int lane = threadIdx.x & 31;
  float v[8]; float mx = -3.4e38f;
  #pragma unroll
  for (int i = 0; i < 8; ++i){
    int e = lane + 32*i;
    int k = e >> 4, j = e & 15;
    const float* ck = emb + (h*16 + k) * 64;
    const float* cj = emb + (h*16 + j) * 64;
    float s = 0.f;
    #pragma unroll
    for (int d = 0; d < 64; ++d) s = fmaf(ck[d], cj[d], s);
    v[i] = s; mx = fmaxf(mx, s);
  }
  #pragma unroll
  for (int off = 16; off > 0; off >>= 1) mx = fmaxf(mx, __shfl_xor(mx, off, 32));
  float sum = 0.f;
  #pragma unroll
  for (int i = 0; i < 8; ++i){ v[i] = __expf(v[i] - mx); sum += v[i]; }
  #pragma unroll
  for (int off = 16; off > 0; off >>= 1) sum += __shfl_xor(sum, off, 32);
  float inv = 1.f / sum;
  #pragma unroll
  for (int i = 0; i < 8; ++i) S[h*256 + lane + 32*i] = v[i] * inv;
}

// ---------------------------------------------------------------------------
// Kernel: 3-layer MLP + sigmoid(· clusters^T); mode 0 -> Qhat, mode 1 -> SKT
// ---------------------------------------------------------------------------
__global__ __launch_bounds__(256) void k_proj(
    const float* __restrict__ X,
    const float* __restrict__ W1, const float* __restrict__ b1,
    const float* __restrict__ W2, const float* __restrict__ b2,
    const float* __restrict__ W3, const float* __restrict__ b3,
    const float* __restrict__ emb, const float* __restrict__ S,
    float* __restrict__ out, int mode){
  __shared__ float sW[3*64*64];
  __shared__ float sC[16*64];
  __shared__ float sB[3*64];
  __shared__ float sS[256];
  int tid = threadIdx.x;
  long long r = (long long)blockIdx.x * 256 + tid;   // 256 | 2048 -> one (b,h) per block
  int h = (int)((r / Nc) % Hc);
  for (int i = tid; i < 64*64; i += 256){
    sW[i] = W1[i]; sW[4096 + i] = W2[i]; sW[8192 + i] = W3[i];
  }
  for (int i = tid; i < 16*64; i += 256) sC[i] = emb[h*16*64 + i];
  if (tid < 64){ sB[tid] = b1[tid]; sB[64+tid] = b2[tid]; sB[128+tid] = b3[tid]; }
  sS[tid] = S[h*256 + tid];
  __syncthreads();

  float x[64];
  const v4f* xp = (const v4f*)(X + r * 64);
  #pragma unroll
  for (int i = 0; i < 16; ++i){
    v4f t = xp[i];
    x[4*i] = t.x; x[4*i+1] = t.y; x[4*i+2] = t.z; x[4*i+3] = t.w;
  }
  float y[64];
  for (int l = 0; l < 3; ++l){
    const float* W = sW + l*4096;
    for (int e = 0; e < 64; ++e){
      float a = sB[l*64 + e];
      const float* w = W + e*64;
      #pragma unroll
      for (int d = 0; d < 64; ++d) a = fmaf(w[d], x[d], a);
      y[e] = (l < 2) ? fmaxf(a, 0.f) : a;
    }
    #pragma unroll
    for (int i = 0; i < 64; ++i) x[i] = y[i];
  }
  float hat[16];
  #pragma unroll
  for (int k = 0; k < 16; ++k){
    const float* c = sC + k*64;
    float t = 0.f;
    #pragma unroll
    for (int d = 0; d < 64; ++d) t = fmaf(c[d], x[d], t);
    hat[k] = 1.f / (1.f + __expf(-t));
  }
  float* o = out + r * 16;
  if (mode == 0){
    #pragma unroll
    for (int k = 0; k < 16; ++k) o[k] = hat[k];
  } else {                                     // SKT[m][k] = sum_j S[k,j] * Khat[m,j]
    #pragma unroll
    for (int k = 0; k < 16; ++k){
      float s = 0.f;
      #pragma unroll
      for (int j = 0; j < 16; ++j) s = fmaf(sS[k*16 + j], hat[j], s);
      o[k] = s;
    }
  }
}

// ---------------------------------------------------------------------------
// Main kernel: one block per (b,h, 16-row n-tile). Full padded score stripe in
// LDS; SKT async-DMA double-buffered; attn written via async LDS->HBM stores.
// ---------------------------------------------------------------------------
__global__ __launch_bounds__(256) void k_attn(
    const int* __restrict__ mask, const float* __restrict__ noise,
    const unsigned short* __restrict__ Qbf, const unsigned short* __restrict__ Kbf,
    const unsigned short* __restrict__ Vt,
    const float* __restrict__ Qhat, const float* __restrict__ SKT,
    float* __restrict__ Xout, float* __restrict__ graph, float* __restrict__ attn,
    float* __restrict__ acc){
  __shared__ float sc[16 * SC_STRIDE];   // 131,328 B score / e / attn stripe
  __shared__ float sSKT[2][128 * 16];    // 16,384 B double-buffered SKT chunks
  __shared__ float sQh[256];
  __shared__ float sRow[16];
  __shared__ float sSum[16];
  __shared__ float sRed[8];
  __shared__ float sX[16 * 64];

  const int tid    = threadIdx.x;
  const int wv     = tid >> 5;
  const int lane   = tid & 31;
  const int ln16   = lane & 15;
  const int hiHalf = lane >> 4;
  const int nt = blockIdx.x & 127;
  const int bh = blockIdx.x >> 7;
  const int b  = bh >> 3;
  const int h  = bh & 7;
  const int n0 = nt * 16;
  const int mbase = wv * 256;          // this wave's 256-column chunk
  const int kbA   = hiHalf * 8;
  const int offB  = hiHalf * 8;

  // ---- Stage A: scores = (Q K^T) / 8 via v_wmma_f32_16x16x32_bf16 ----
  const unsigned short* qrow = Qbf + ((long long)(bh*Nc + n0 + ln16)) * 64;
  v16bf a0 = load_bf_A(qrow, kbA);          // d = 0..31
  v16bf a1 = load_bf_A(qrow, 32 + kbA);     // d = 32..63
  // single base pointer: tile offsets are compile-time constants (2KB apart)
  const unsigned short* kbase = Kbf + ((long long)(bh*Mc + mbase + ln16)) * 64 + offB;
  const int* mrow = mask + b*Mc + mbase + ln16;
  #pragma unroll 4
  for (int t = 0; t < 16; ++t){
    v16bf b0 = load_bf_B(kbase + t*1024, 0);
    v16bf b1 = load_bf_B(kbase + t*1024, 32);
    v8f c = {};
    c = __builtin_amdgcn_wmma_f32_16x16x32_bf16(false, a0, false, b0, (short)0, c, false, false);
    c = __builtin_amdgcn_wmma_f32_16x16x32_bf16(false, a1, false, b1, (short)0, c, false, false);
    int mk = mrow[t*16];
    #pragma unroll
    for (int r2 = 0; r2 < 8; ++r2){
      int mm = r2 + hiHalf*8;               // D layout: lane<16 -> rows 0..7, else 8..15
      sc[mm*SC_STRIDE + mbase + t*16 + ln16] = mk ? -3.0e38f : c[r2] * 0.125f;
    }
  }
  __syncthreads();

  // ---- Stage B: row max (2 rows per wave) ----
  #pragma unroll
  for (int rr = 0; rr < 2; ++rr){
    int row = wv*2 + rr;
    float mx = -3.4e38f;
    for (int i = lane; i < 2048; i += 32) mx = fmaxf(mx, sc[row*SC_STRIDE + i]);
    #pragma unroll
    for (int off = 16; off > 0; off >>= 1) mx = fmaxf(mx, __shfl_xor(mx, off, 32));
    if (lane == 0) sRow[row] = mx;
  }
  #pragma unroll
  for (int j = 0; j < 4; ++j) sX[tid + 256*j] = 0.f;

  // ---- Stage C: expA via LDS-staged SKT (async DMA), graph, e, row sums ----
  {
    const int row = tid >> 4;
    const int q   = tid & 15;
    sQh[tid] = Qhat[((long long)(bh*Nc + n0 + row)) * 16 + q];

    const float*   sktg = SKT + ((long long)bh * Mc) * 16;       // (b,h) slab base
    const unsigned lds0 = (unsigned)(uintptr_t)(&sSKT[0][0]);
    const unsigned toff = (unsigned)(tid * 16);

    // prologue: async-stage chunk 0 into buffer 0 (8KB = 2 x 4KB rounds)
    asm volatile("global_load_async_to_lds_b128 %0, %1, %2"
                 :: "v"(lds0 + toff),        "v"(toff),        "s"(sktg) : "memory");
    asm volatile("global_load_async_to_lds_b128 %0, %1, %2"
                 :: "v"(lds0 + toff + 4096), "v"(toff + 4096), "s"(sktg) : "memory");

    __syncthreads();                    // sRow / sQh / sX-zero visible
    float qh[16];
    #pragma unroll
    for (int k = 0; k < 16; ++k) qh[k] = sQh[row*16 + k];

    float rmax = sRow[row];
    long long nb = ((long long)(bh*Nc + n0 + row)) * Mc;
    float psum = 0.f, gcnt = 0.f;

    for (int j = 0; j < 16; ++j){       // 16 chunks of 128 columns
      int c0 = j << 7;
      if (j < 15){                      // stage chunk j+1 into the other buffer
        unsigned goff = (unsigned)((c0 + 128) * 64) + toff;
        unsigned loff = lds0 + (unsigned)(((j + 1) & 1) * 8192) + toff;
        asm volatile("global_load_async_to_lds_b128 %0, %1, %2"
                     :: "v"(loff),        "v"(goff),        "s"(sktg) : "memory");
        asm volatile("global_load_async_to_lds_b128 %0, %1, %2"
                     :: "v"(loff + 4096), "v"(goff + 4096), "s"(sktg) : "memory");
        asm volatile("s_wait_asynccnt 0x2" ::: "memory");   // chunk j complete
        __builtin_prefetch(noise + nb + c0 + 128 + (q << 2), 0, 1);
      } else {
        asm volatile("s_wait_asynccnt 0x0" ::: "memory");
      }
      __syncthreads();                  // chunk j resident for all waves
      const float* sk = &sSKT[j & 1][0];
      #pragma unroll
      for (int g = 0; g < 2; ++g){      // 2 groups of 4 consecutive columns
        int ccb = (g << 6) + (q << 2);
        int col = c0 + ccb;
        long long gi = nb + col;
        v4f sv = *(v4f*)&sc[row*SC_STRIDE + col];
        v4f nz = *(const v4f*)&noise[gi];
        float ea4[4];
        #pragma unroll
        for (int e = 0; e < 4; ++e){
          const float* skc = sk + (ccb + e) * 16;
          float ea = 0.f;
          #pragma unroll
          for (int j2 = 0; j2 < 4; ++j2){
            int rot = (j2 + q) & 3;     // rotate sub-vectors -> spread LDS banks
            v4f t = *(const v4f*)(skc + rot*4);
            ea = fmaf(qh[4*rot],   t.x, ea); ea = fmaf(qh[4*rot+1], t.y, ea);
            ea = fmaf(qh[4*rot+2], t.z, ea); ea = fmaf(qh[4*rot+3], t.w, ea);
          }
          ea4[e] = ea;
        }
        v4f gv;
        gv.x = (nz.x < ea4[0]) ? 1.f : 0.f;
        gv.y = (nz.y < ea4[1]) ? 1.f : 0.f;
        gv.z = (nz.z < ea4[2]) ? 1.f : 0.f;
        gv.w = (nz.w < ea4[3]) ? 1.f : 0.f;
        *(v4f*)&graph[gi] = gv;
        gcnt += gv.x + gv.y + gv.z + gv.w;
        v4f ev;
        ev.x = (sv.x <= -1.0e30f) ? 0.f : __expf(sv.x - rmax) * gv.x;
        ev.y = (sv.y <= -1.0e30f) ? 0.f : __expf(sv.y - rmax) * gv.y;
        ev.z = (sv.z <= -1.0e30f) ? 0.f : __expf(sv.z - rmax) * gv.z;
        ev.w = (sv.w <= -1.0e30f) ? 0.f : __expf(sv.w - rmax) * gv.w;
        *(v4f*)&sc[row*SC_STRIDE + col] = ev;
        psum += ev.x + ev.y + ev.z + ev.w;
      }
      __syncthreads();                  // done reading buf before it is re-staged
    }

    #pragma unroll
    for (int off = 8; off > 0; off >>= 1) psum += __shfl_xor(psum, off, 16);
    if (q == 0) sSum[row] = psum;
    #pragma unroll
    for (int off = 16; off > 0; off >>= 1) gcnt += __shfl_xor(gcnt, off, 32);
    if (lane == 0) sRed[wv] = gcnt;
  }
  __syncthreads();
  if (tid == 0){
    float tot = 0.f;
    #pragma unroll
    for (int i = 0; i < 8; ++i) tot += sRed[i];
    atomicAdd(acc + h, tot);                 // integer-valued -> exact in f32 (< 2^24)
  }

  // ---- Stage D(a): normalize sc in place (sc becomes attn) ----
  {
    int row = tid >> 4, q = tid & 15;
    float inv = 1.f / fmaxf(sSum[row], 1e-12f);
    for (int i = 0; i < 32; ++i){
      int col = i*64 + q*4;
      v4f* p = (v4f*)&sc[row*SC_STRIDE + col];
      v4f v = *p;
      v.x *= inv; v.y *= inv; v.z *= inv; v.w *= inv;
      *p = v;
    }
  }
  __syncthreads();

  // ---- Stage D(b): async DMA attn rows LDS -> HBM (overlaps WMMA below) ----
  {
    const float*   attnB  = attn + ((long long)(bh*Nc + n0)) * Mc;
    const unsigned scLds0 = (unsigned)(uintptr_t)(&sc[0]);
    #pragma unroll
    for (int rr = 0; rr < 2; ++rr){
      int row = wv*2 + rr;
      unsigned lbase = scLds0 + (unsigned)(row * SC_STRIDE * 4) + (unsigned)(lane * 16);
      unsigned gbase = (unsigned)(row * Mc * 4) + (unsigned)(lane * 16);
      for (int i = 0; i < 16; ++i){     // 16 x 512B per row
        asm volatile("global_store_async_from_lds_b128 %0, %1, %2"
                     :: "v"(gbase + i*512), "v"(lbase + i*512), "s"(attnB) : "memory");
      }
    }
  }

  // ---- Stage D(c): X = attn @ V via WMMA against pre-transposed Vt [bh,d,M] ----
  {
    v8f xc[4];
    #pragma unroll
    for (int dt = 0; dt < 4; ++dt) xc[dt] = (v8f){};
    const unsigned short* vbase = Vt + ((long long)(bh*64 + ln16)) * Mc + mbase + offB;
    for (int ks = 0; ks < 8; ++ks){
      int moff = mbase + ks*32;
      // A operand: attn-tile rows from LDS (padded stride -> conflict-free b128)
      const float* srow = &sc[ln16*SC_STRIDE + moff + kbA];
      v4f f0 = *(const v4f*)(srow);
      v4f f1 = *(const v4f*)(srow + 4);
      v4f f2 = *(const v4f*)(srow + 16);
      v4f f3 = *(const v4f*)(srow + 20);
      ABPack pa; pa.q[0] = pack8(f0, f1); pa.q[1] = pack8(f2, f3);
      v16bf av = pa.v;
      #pragma unroll
      for (int dt = 0; dt < 4; ++dt){   // Vt slice offsets are constants (64KB apart)
        v16bf bv = load_bf_B(vbase + dt*(16*Mc) + ks*32, 0);
        xc[dt] = __builtin_amdgcn_wmma_f32_16x16x32_bf16(false, av, false, bv,
                                                         (short)0, xc[dt], false, false);
      }
    }
    #pragma unroll
    for (int dt = 0; dt < 4; ++dt){
      #pragma unroll
      for (int r2 = 0; r2 < 8; ++r2){
        int mm = r2 + hiHalf*8;
        atomicAdd(&sX[mm*64 + dt*16 + ln16], xc[dt][r2]);   // ds_add_f32 cross-wave reduce
      }
    }
  }
  __syncthreads();

  // ---- Stage D(d): write X (already normalized: sc held attn) ----
  #pragma unroll
  for (int j = 0; j < 4; ++j){
    int idx = tid + 256*j;
    int mm = idx >> 6, d = idx & 63;
    Xout[((long long)(bh*Nc + n0 + mm)) * 64 + d] = sX[idx];
  }
}

// ---------------------------------------------------------------------------
// Finalize sparsity
// ---------------------------------------------------------------------------
__global__ void k_final(const float* __restrict__ acc, float* __restrict__ sp){
  int t = threadIdx.x;
  if (t < 8) sp[t] = acc[t] * (1.f / (float)((long long)Bc * Nc * Mc));
}

// ---------------------------------------------------------------------------
// Host launcher
// ---------------------------------------------------------------------------
extern "C" void kernel_launch(void* const* d_in, const int* in_sizes, int n_in,
                              void* d_out, int out_size, void* d_ws, size_t ws_size,
                              hipStream_t stream){
  (void)in_sizes; (void)n_in; (void)out_size; (void)ws_size;
  const float* Q    = (const float*)d_in[0];
  const float* K    = (const float*)d_in[1];
  const float* V    = (const float*)d_in[2];
  const int*   mask = (const int*)  d_in[3];
  const float* noise= (const float*)d_in[4];
  const float* emb  = (const float*)d_in[5];
  const float* W1 = (const float*)d_in[6];  const float* b1 = (const float*)d_in[7];
  const float* W2 = (const float*)d_in[8];  const float* b2 = (const float*)d_in[9];
  const float* W3 = (const float*)d_in[10]; const float* b3 = (const float*)d_in[11];

  // workspace layout (bytes), 256-aligned slots; total ~16.8 MB
  char* ws = (char*)d_ws;
  float* Sw   = (float*)(ws);                                   // 8,192
  float* acc  = (float*)(ws + 8192);                            // 256
  float* Qhat = (float*)(ws + 8448);                            // 2,097,152
  float* SKT  = (float*)(ws + 8448 + 2097152);                  // 2,097,152
  unsigned short* Qbf = (unsigned short*)(ws + 8448 + 2*2097152);             // 4,194,304
  unsigned short* Kbf = (unsigned short*)(ws + 8448 + 2*2097152 + 4194304);   // 4,194,304
  unsigned short* Vt  = (unsigned short*)(ws + 8448 + 2*2097152 + 2*4194304); // 4,194,304

  float* Xout  = (float*)d_out;
  float* spar  = Xout + XSZ;
  float* graph = spar + 8;
  float* attn  = graph + GSZ;

  k_prep    <<<(unsigned)((XSZ + 255) / 256), 256, 0, stream>>>(Q, K, V, Qbf, Kbf, Vt, acc);
  k_clusterS<<<1, 256, 0, stream>>>(emb, Sw);
  k_proj    <<<BHc * Nc / 256, 256, 0, stream>>>(Q, W1, b1, W2, b2, W3, b3, emb, Sw, Qhat, 0);
  k_proj    <<<BHc * Mc / 256, 256, 0, stream>>>(K, W1, b1, W2, b2, W3, b3, emb, Sw, SKT, 1);
  k_attn    <<<BHc * (Nc / 16), 256, 0, stream>>>(mask, noise, Qbf, Kbf, Vt, Qhat, SKT,
                                                  Xout, graph, attn, acc);
  k_final   <<<1, 64, 0, stream>>>(acc, spar);
}
